// hybrid_Decoder_64312840290455
// MI455X (gfx1250) — matrix operand order repaired and checked by reading, not compiled, not run on hardware
//
#include <hip/hip_runtime.h>

typedef __attribute__((ext_vector_type(16))) _Float16 v16h;
typedef __attribute__((ext_vector_type(8)))  _Float16 v8h;
typedef __attribute__((ext_vector_type(8)))  float    v8f;

#define BLK   128          // 4 waves of 32
#define WAVES 4
#define KPAD  128          // 98-feature rows padded to 128 (4 K-chunks of 32)
#define FS    136          // feature row stride in f16 (272B, 16B aligned, bank-skewed)
#define BS    136          // B^T row stride in f16
#define HS    40           // hidden / B2^T row stride in f16 (80B)
#define RES   512
#define HWSZ  (RES*RES)

struct Bil { int i00, i01, i10, i11; float w00, w01, w10, w11; };

__device__ __forceinline__ Bil make_bil(float gx, float gy) {
    float px = fminf(fmaxf((gx + 1.0f) * 0.5f * (float)(RES - 1), 0.0f), (float)(RES - 1));
    float py = fminf(fmaxf((gy + 1.0f) * 0.5f * (float)(RES - 1), 0.0f), (float)(RES - 1));
    float x0 = floorf(px), y0 = floorf(py);
    float wx = px - x0,   wy = py - y0;
    int x0i = (int)x0, y0i = (int)y0;
    int x1i = min(x0i + 1, RES - 1), y1i = min(y0i + 1, RES - 1);
    Bil b;
    b.i00 = y0i * RES + x0i; b.i01 = y0i * RES + x1i;
    b.i10 = y1i * RES + x0i; b.i11 = y1i * RES + x1i;
    b.w00 = (1.0f - wy) * (1.0f - wx); b.w01 = (1.0f - wy) * wx;
    b.w10 = wy * (1.0f - wx);          b.w11 = wy * wx;
    return b;
}

__device__ __forceinline__ v16h ld_a(const _Float16* p0, const _Float16* p1) {
    v8h lo = *(const v8h*)p0;
    v8h hi = *(const v8h*)p1;
    v16h a;
#pragma unroll
    for (int i = 0; i < 8; ++i) { a[i] = lo[i]; a[8 + i] = hi[i]; }
    return a;
}

// layer0: D[16x32] += A[16x128] @ B[128x32]  as 4 K-chunks x 2 N-tiles of wmma f16
__device__ __forceinline__ void layer0(const _Float16* Bt, const v16h* A,
                                       int half, int nn, v8f acc[2]) {
#pragma unroll
    for (int c = 0; c < 4; ++c) {
#pragma unroll
        for (int nt = 0; nt < 2; ++nt) {
            const _Float16* brow = Bt + (nt * 16 + nn) * BS + c * 32 + 8 * half;
            v16h b = ld_a(brow, brow + 16);
            acc[nt] = __builtin_amdgcn_wmma_f32_16x16x32_f16(
                false, A[c], false, b, (short)0, acc[nt], false, false);
        }
    }
}

__global__ __launch_bounds__(BLK)
void triplane_decoder(const float* __restrict__ qp,   const float* __restrict__ bbox,
                      const float* __restrict__ pxy,  const float* __restrict__ pxz,
                      const float* __restrict__ pyz,
                      const float* __restrict__ wsdf0, const float* __restrict__ wsdf1,
                      const float* __restrict__ wcol0, const float* __restrict__ wcol1,
                      float* __restrict__ out, int npts)
{
    __shared__ __align__(16) _Float16 sFeat[BLK * FS];       // per-point 128-K f16 rows
    __shared__ __align__(16) _Float16 sBsdf[32 * BS];        // w_sdf0^T  [N=32][K=128]
    __shared__ __align__(16) _Float16 sBcol[32 * BS];        // w_col0^T, K-permuted to sdf layout
    __shared__ __align__(16) _Float16 sB2sdf[16 * HS];       // [N=16][K=32], col 0 = w_sdf1
    __shared__ __align__(16) _Float16 sB2col[16 * HS];       // cols 0..2 = w_col1
    __shared__ __align__(16) _Float16 sHid[WAVES * 16 * HS]; // per-wave hidden relayout buffer

    const int tid = threadIdx.x;

    // ---- weights -> LDS as f16 B^T layouts (zero padded / permuted) ----
    for (int idx = tid; idx < 32 * KPAD; idx += BLK) {
        int n = idx >> 7, k = idx & 127;
        sBsdf[n * BS + k] = (_Float16)((k < 98) ? wsdf0[k * 32 + n] : 0.0f);
        // col K order matches sdf feature rows: [geo 0..26 | feat 27..58 | pos 59..97]
        float vc = 0.0f;
        if (k < 27)                vc = wcol0[(39 + k) * 32 + n];  // geo part of col input
        else if (k >= 59 && k < 98) vc = wcol0[(k - 59) * 32 + n]; // pos part of col input
        sBcol[n * BS + k] = (_Float16)vc;
    }
    for (int idx = tid; idx < 16 * 32; idx += BLK) {
        int n = idx >> 5, k = idx & 31;
        sB2sdf[n * HS + k] = (_Float16)((n == 0) ? wsdf1[k] : 0.0f);
        sB2col[n * HS + k] = (_Float16)((n < 3) ? wcol1[k * 3 + n] : 0.0f);
    }

    // ---- per-thread: normalize, freq-embed, tri-plane bilinear gather ----
    int pid = blockIdx.x * BLK + tid;
    if (pid >= npts) pid = npts - 1;

    float lo0 = bbox[0], hi0 = bbox[1];
    float lo1 = bbox[2], hi1 = bbox[3];
    float lo2 = bbox[4], hi2 = bbox[5];
    float pn[3];
    pn[0] = (qp[pid * 3 + 0] - lo0) / (hi0 - lo0);
    pn[1] = (qp[pid * 3 + 1] - lo1) / (hi1 - lo1);
    pn[2] = (qp[pid * 3 + 2] - lo2) / (hi2 - lo2);

    _Float16* row = &sFeat[tid * FS];
    // geo embed: [p(3) | sin d*4+l (3..14) | cos (15..26)]
    row[0] = (_Float16)pn[0]; row[1] = (_Float16)pn[1]; row[2] = (_Float16)pn[2];
#pragma unroll
    for (int d = 0; d < 3; ++d) {
        float f = 1.0f;
#pragma unroll
        for (int l = 0; l < 4; ++l) {
            float a = pn[d] * f;
            row[3 + d * 4 + l]  = (_Float16)__sinf(a);
            row[15 + d * 4 + l] = (_Float16)__cosf(a);
            f *= 2.0f;
        }
    }
    // pos embed at K=59: [p(3) | sin 62 + d*6+l | cos 80 + d*6+l]
    row[59] = (_Float16)pn[0]; row[60] = (_Float16)pn[1]; row[61] = (_Float16)pn[2];
#pragma unroll
    for (int d = 0; d < 3; ++d) {
        float f = 1.0f;
#pragma unroll
        for (int l = 0; l < 6; ++l) {
            float a = pn[d] * f;
            row[62 + d * 6 + l] = (_Float16)__sinf(a);
            row[80 + d * 6 + l] = (_Float16)__cosf(a);
            f *= 2.0f;
        }
    }
#pragma unroll
    for (int k = 98; k < KPAD; ++k) row[k] = (_Float16)0.0f;

    // tri-plane gather (planes total 96MB -> resident in 192MB L2)
    Bil bxy = make_bil(pn[0], pn[1]);
    Bil bxz = make_bil(pn[0], pn[2]);
    Bil byz = make_bil(pn[1], pn[2]);
#pragma unroll 8
    for (int c = 0; c < 32; ++c) {
        const float* a0 = pxy + c * HWSZ;
        const float* a1 = pxz + c * HWSZ;
        const float* a2 = pyz + c * HWSZ;
        float f = a0[bxy.i00] * bxy.w00 + a0[bxy.i01] * bxy.w01 +
                  a0[bxy.i10] * bxy.w10 + a0[bxy.i11] * bxy.w11 +
                  a1[bxz.i00] * bxz.w00 + a1[bxz.i01] * bxz.w01 +
                  a1[bxz.i10] * bxz.w10 + a1[bxz.i11] * bxz.w11 +
                  a2[byz.i00] * byz.w00 + a2[byz.i01] * byz.w01 +
                  a2[byz.i10] * byz.w10 + a2[byz.i11] * byz.w11;
        row[27 + c] = (_Float16)f;
    }

    __syncthreads();   // weights + all feature rows visible

    // ---- WMMA phase: each wave owns its 32 points (2 x 16-row tiles) ----
    const int lane = tid & 31;
    const int wav  = tid >> 5;
    const int half = lane >> 4;
    const int nn   = lane & 15;
    _Float16* hid  = &sHid[wav * 16 * HS];
    const int blkbase = blockIdx.x * BLK;

#pragma unroll
    for (int mt = 0; mt < 2; ++mt) {
        const int rowbase = wav * 32 + mt * 16;
        const _Float16* arow = &sFeat[(rowbase + nn) * FS];

        // A chunks shared by both MLPs (col weights were K-permuted to match)
        v16h A[4];
#pragma unroll
        for (int c = 0; c < 4; ++c)
            A[c] = ld_a(arow + c * 32 + 8 * half, arow + c * 32 + 16 + 8 * half);

        const int ptbase = blkbase + rowbase;

        // ---------- SDF MLP ----------
        {
            v8f acc[2] = {};
            layer0(sBsdf, A, half, nn, acc);
#pragma unroll
            for (int nt = 0; nt < 2; ++nt)
#pragma unroll
                for (int r = 0; r < 8; ++r)
                    hid[(r + 8 * half) * HS + nt * 16 + nn] =
                        (_Float16)fmaxf(acc[nt][r], 0.0f);

            asm volatile("s_wait_dscnt 0" ::: "memory");  // cross-lane hidden relayout

            v16h a2 = ld_a(hid + nn * HS + 8 * half, hid + nn * HS + 16 + 8 * half);
            v16h b2 = ld_a(sB2sdf + nn * HS + 8 * half, sB2sdf + nn * HS + 8 * half + 16);
            v8f z = {};
            v8f d = __builtin_amdgcn_wmma_f32_16x16x32_f16(
                false, a2, false, b2, (short)0, z, false, false);
            if (nn == 0) {
#pragma unroll
                for (int r = 0; r < 8; ++r) {
                    int pt = ptbase + r + 8 * half;
                    if (pt < npts) out[pt * 4 + 3] = d[r];
                }
            }
        }

        // ---------- color MLP (reuses per-wave hidden buffer) ----------
        {
            v8f acc[2] = {};
            layer0(sBcol, A, half, nn, acc);
            asm volatile("s_wait_dscnt 0" ::: "memory");  // sdf a2 loads done before overwrite
#pragma unroll
            for (int nt = 0; nt < 2; ++nt)
#pragma unroll
                for (int r = 0; r < 8; ++r)
                    hid[(r + 8 * half) * HS + nt * 16 + nn] =
                        (_Float16)fmaxf(acc[nt][r], 0.0f);

            asm volatile("s_wait_dscnt 0" ::: "memory");

            v16h a2 = ld_a(hid + nn * HS + 8 * half, hid + nn * HS + 16 + 8 * half);
            v16h b2 = ld_a(sB2col + nn * HS + 8 * half, sB2col + nn * HS + 8 * half + 16);
            v8f z = {};
            v8f d = __builtin_amdgcn_wmma_f32_16x16x32_f16(
                false, a2, false, b2, (short)0, z, false, false);
            if (nn < 3) {
#pragma unroll
                for (int r = 0; r < 8; ++r) {
                    int pt = ptbase + r + 8 * half;
                    if (pt < npts) out[pt * 4 + nn] = d[r];
                }
            }
        }
    }
}

extern "C" void kernel_launch(void* const* d_in, const int* in_sizes, int n_in,
                              void* d_out, int out_size, void* d_ws, size_t ws_size,
                              hipStream_t stream) {
    const float* qp    = (const float*)d_in[0];
    const float* bbox  = (const float*)d_in[1];
    const float* pxy   = (const float*)d_in[2];
    const float* pxz   = (const float*)d_in[3];
    const float* pyz   = (const float*)d_in[4];
    const float* wsdf0 = (const float*)d_in[5];
    const float* wsdf1 = (const float*)d_in[6];
    const float* wcol0 = (const float*)d_in[7];
    const float* wcol1 = (const float*)d_in[8];
    float* out = (float*)d_out;

    int npts = in_sizes[0] / 3;
    int blocks = (npts + BLK - 1) / BLK;
    triplane_decoder<<<blocks, BLK, 0, stream>>>(
        qp, bbox, pxy, pxz, pyz, wsdf0, wsdf1, wcol0, wcol1, out, npts);
}
